// ACTModelWithPonderTracking_19327352832175
// MI455X (gfx1250) — compile-verified, matrix operand order checked by reading, not run on hardware
//
#include <hip/hip_runtime.h>

// ---------------- problem constants (from reference) ----------------
#define I_SZ   64
#define H_SZ   512
#define M_STEPS 5
#define B_SZ   128
#define T_SZ   256
#define KREAL  577          // 64 x + 1 flag + 512 h
#define KDIM   640          // padded to 20*32: two K-halves of exactly 10 wmma steps
#define KHALF  10           // k-steps per K-half (compile-time constant!)
#define EPS_ACT 0.01f

typedef __attribute__((ext_vector_type(16))) __bf16 v16bf;
typedef __attribute__((ext_vector_type(8)))  __bf16 v8bf;
typedef __attribute__((ext_vector_type(8)))  float  v8f;

__device__ __forceinline__ float sigmoidf_(float x) {
    return 1.0f / (1.0f + __expf(-x));
}

#define CAT16(lo, hi) __builtin_shufflevector((lo), (hi), 0,1,2,3,4,5,6,7,8,9,10,11,12,13,14,15)

// ---------------------------------------------------------------
// Build WcatT[n][k] (bf16, row-major over n=0..2047, k=0..639):
//   k <  65   : W_ih[n][k]        (includes flag column at k==64)
//   65<=k<577 : W_hh[n][k-65]
//   else      : 0 (pad)
// ---------------------------------------------------------------
__global__ void build_wcat_kernel(const float* __restrict__ W_ih,
                                  const float* __restrict__ W_hh,
                                  __bf16* __restrict__ WcatT) {
    int n = blockIdx.x;                       // 0..2047
    for (int k = threadIdx.x; k < KDIM; k += blockDim.x) {
        float v = 0.0f;
        if (k < I_SZ + 1)      v = W_ih[(size_t)n * (I_SZ + 1) + k];
        else if (k < KREAL)    v = W_hh[(size_t)n * H_SZ + (k - (I_SZ + 1))];
        WcatT[(size_t)n * KDIM + k] = (__bf16)v;
    }
}

// Zero h_state, c_state, hdot, ponder accumulator.
__global__ void init_state_kernel(float* __restrict__ h_state,
                                  float* __restrict__ c_state,
                                  float* __restrict__ hdot,
                                  float* __restrict__ ponder_acc) {
    int idx = blockIdx.x * blockDim.x + threadIdx.x;
    if (idx < B_SZ * H_SZ) { h_state[idx] = 0.0f; c_state[idx] = 0.0f; }
    if (idx < M_STEPS * B_SZ) hdot[idx] = 0.0f;
    if (idx == 0) ponder_acc[0] = 0.0f;
}

// ---------------------------------------------------------------
// Pack activation matrix Act[b][k] (bf16, row-major, K padded):
//   k<64 : x[b][t][k], k==64 : flag, 65<=k<577 : h_src[b][k-65], pad 0
// For ponder steps n>=1 (full==0) only the flag/h region changes, so
// the x part is left untouched. Also zeroes this step's hdot_n[b].
// ---------------------------------------------------------------
__global__ void pack_act_kernel(const float* __restrict__ x,
                                const float* __restrict__ h_src,
                                __bf16* __restrict__ Act,
                                float* __restrict__ hdot_n,
                                int t, float flag, int full) {
    int b = blockIdx.x;                       // 0..127
    const float* xr = x + ((size_t)b * T_SZ + t) * I_SZ;
    const int k0 = full ? 0 : I_SZ;           // skip invariant x region when possible
    for (int k = k0 + threadIdx.x; k < KDIM; k += blockDim.x) {
        float v;
        if (k < I_SZ)          v = xr[k];
        else if (k == I_SZ)    v = flag;
        else if (k < KREAL)    v = h_src[(size_t)b * H_SZ + (k - (I_SZ + 1))];
        else                   v = 0.0f;
        Act[(size_t)b * KDIM + k] = (__bf16)v;
    }
    if (threadIdx.x == 0) hdot_n[b] = 0.0f;
}

// ---------------------------------------------------------------
// Fused gates-GEMM + LSTM pointwise + halting partial reduction.
// Block = 256 threads = 8 waves; block (mt,ct) owns a 16x16 (batch x h)
// tile.  Wave w = g*2+kh computes gate g over K-half kh: a fully
// unrolled, constant-trip-count chain of KHALF=10 wmma_f32_16x16x32_bf16.
// Partials are reduced through LDS and the 256 threads do the LSTM
// pointwise (one (row,col) element each) + halting dot reduction.
// ---------------------------------------------------------------
__global__ __launch_bounds__(256)
void lstm_step_kernel(const __bf16* __restrict__ Act,
                      const __bf16* __restrict__ WcatT,
                      const float*  __restrict__ bias,
                      const float*  __restrict__ c_src,
                      const float*  __restrict__ w_halt,
                      float* __restrict__ Hn,
                      float* __restrict__ Cn,
                      float* __restrict__ hdot_n) {
    const int tid  = threadIdx.x;
    const int lane = tid & 31;
    const int w    = tid >> 5;                // wave 0..7
    const int g    = w >> 1;                  // gate 0..3
    const int kh   = w & 1;                   // K half 0..1
    const int mt   = blockIdx.x;              // 0..7   (batch row tile)
    const int ct   = blockIdx.y;              // 0..31  (h column tile)

    __shared__ float partial[8 * 32 * 8];     // [wave][lane][r] = 8 KB
    __shared__ float hred[16];
    if (tid < 16) hred[tid] = 0.0f;

    // 16-bit operand lane layout (ISA 7.12.2): lanes 0-15 hold row/col L,
    // K in {0..7,16..23}; lanes 16-31 hold the same rows, K in {8..15,24..31}.
    const int rsel = lane & 15;
    const int koff = (lane >> 4) * 8;
    const int kofs = kh * (KHALF * 32);       // K-half = pointer offset, not a bound

    const __bf16* aP = Act + (size_t)(mt * 16 + rsel) * KDIM + kofs + koff;
    const __bf16* bP = WcatT + (size_t)(g * H_SZ + ct * 16 + rsel) * KDIM + kofs + koff;

    v8f acc = {};
    #pragma unroll
    for (int kk = 0; kk < KHALF; ++kk) {      // constant bounds -> full unroll
        const int kb = kk * 32;
        v16bf a = CAT16(*(const v8bf*)(aP + kb), *(const v8bf*)(aP + kb + 16));
        v16bf m = CAT16(*(const v8bf*)(bP + kb), *(const v8bf*)(bP + kb + 16));
        acc = __builtin_amdgcn_wmma_f32_16x16x32_bf16(false, a, false, m, (short)0, acc, false, false);
    }

    // stash this wave's partial accumulator: [w][lane][r], 32B per lane
    *(v8f*)&partial[(w * 32 + lane) * 8] = acc;
    __syncthreads();

    // ---- pointwise: one thread per element of the 16x16 tile ----
    // C/D layout: element (m, n) lives in VGPR m%8, lane (m/8)*16 + n.
    const int row  = tid >> 4;                // 0..15 within tile
    const int coll = tid & 15;                // 0..15 within tile
    const int r    = row & 7;
    const int clane = (row >> 3) * 16 + coll;

    float gate[4];
    #pragma unroll
    for (int gg = 0; gg < 4; ++gg) {
        gate[gg] = partial[((gg * 2 + 0) * 32 + clane) * 8 + r]
                 + partial[((gg * 2 + 1) * 32 + clane) * 8 + r];
    }

    const int col  = ct * 16 + coll;          // h index within H
    const int grow = mt * 16 + row;           // batch index
    const float iv = gate[0] + bias[0 * H_SZ + col];
    const float fv = gate[1] + bias[1 * H_SZ + col];
    const float gv = gate[2] + bias[2 * H_SZ + col];
    const float ov = gate[3] + bias[3 * H_SZ + col];
    const float cp = c_src[(size_t)grow * H_SZ + col];
    const float cn = sigmoidf_(fv) * cp + sigmoidf_(iv) * tanhf(gv);
    const float hn = sigmoidf_(ov) * tanhf(cn);
    Cn[(size_t)grow * H_SZ + col] = cn;
    Hn[(size_t)grow * H_SZ + col] = hn;

    atomicAdd(&hred[row], hn * w_halt[col]);  // ds_add_f32 partial halt dot
    __syncthreads();
    if (tid < 16) atomicAdd(&hdot_n[mt * 16 + tid], hred[tid]);
}

// ---------------------------------------------------------------
// ACT combine: compute halting probs from hdot, weight the M ponder
// states into h_state/c_state, accumulate ponder cost.
// ---------------------------------------------------------------
__global__ void act_combine_kernel(const float* __restrict__ Hs,   // [M][B][H]
                                   const float* __restrict__ Cs,   // [M][B][H]
                                   const float* __restrict__ hdot, // [M][B]
                                   const float* __restrict__ b_halt,
                                   float* __restrict__ h_state,
                                   float* __restrict__ c_state,
                                   float* __restrict__ ponder_acc) {
    const int idx = blockIdx.x * blockDim.x + threadIdx.x;  // B*H threads
    const int b = idx >> 9;
    const int h = idx & (H_SZ - 1);
    const float bh = b_halt[0];

    float w[M_STEPS];
    float cum = 0.0f, nsteps = 0.0f, rem = 0.0f;
    #pragma unroll
    for (int n = 0; n < M_STEPS; ++n) {
        const float p = sigmoidf_(hdot[n * B_SZ + b] + bh);
        const float prev = cum;
        cum += p;
        const bool running  = prev < (1.0f - EPS_ACT);
        const bool halt_now = running && ((cum >= 1.0f - EPS_ACT) || (n == M_STEPS - 1));
        w[n] = running ? (halt_now ? (1.0f - prev) : p) : 0.0f;
        nsteps += running ? 1.0f : 0.0f;
        rem    += halt_now ? (1.0f - prev) : 0.0f;
    }

    float ho = 0.0f, co = 0.0f;
    #pragma unroll
    for (int n = 0; n < M_STEPS; ++n) {
        const size_t off = (size_t)n * B_SZ * H_SZ + (size_t)b * H_SZ + h;
        ho += w[n] * Hs[off];
        co += w[n] * Cs[off];
    }
    h_state[(size_t)b * H_SZ + h] = ho;
    c_state[(size_t)b * H_SZ + h] = co;

    if (h == 0) atomicAdd(ponder_acc, nsteps + rem);
}

// ---------------------------------------------------------------
// Final FC head + ponder scalar.  out[0..127] = h @ W_fc^T + b_fc,
// out[128] = mean ponder.
// ---------------------------------------------------------------
__global__ void fc_out_kernel(const float* __restrict__ h_state,
                              const float* __restrict__ W_fc,
                              const float* __restrict__ b_fc,
                              const float* __restrict__ ponder_acc,
                              float* __restrict__ out) {
    const int b = blockIdx.x;                 // 0..127
    __shared__ float red[256];
    float s = 0.0f;
    for (int h = threadIdx.x; h < H_SZ; h += blockDim.x)
        s += h_state[(size_t)b * H_SZ + h] * W_fc[h];
    red[threadIdx.x] = s;
    __syncthreads();
    for (int off = 128; off > 0; off >>= 1) {
        if (threadIdx.x < off) red[threadIdx.x] += red[threadIdx.x + off];
        __syncthreads();
    }
    if (threadIdx.x == 0) {
        out[b] = red[0] + b_fc[0];
        if (b == 0) out[B_SZ] = ponder_acc[0] * (1.0f / (float)(B_SZ * T_SZ));
    }
}

// ---------------------------------------------------------------
extern "C" void kernel_launch(void* const* d_in, const int* in_sizes, int n_in,
                              void* d_out, int out_size, void* d_ws, size_t ws_size,
                              hipStream_t stream) {
    const float* x      = (const float*)d_in[0];
    const float* W_ih   = (const float*)d_in[1];
    const float* W_hh   = (const float*)d_in[2];
    const float* bias   = (const float*)d_in[3];
    const float* w_halt = (const float*)d_in[4];
    const float* b_halt = (const float*)d_in[5];
    const float* W_fc   = (const float*)d_in[6];
    const float* b_fc   = (const float*)d_in[7];
    float* out = (float*)d_out;

    // ---- workspace carve-up (256 B aligned) ----
    char* p = (char*)d_ws;
    auto take = [&](size_t bytes) {
        char* r = p;
        p += (bytes + 255) & ~(size_t)255;
        return r;
    };
    __bf16* WcatT   = (__bf16*)take((size_t)4 * H_SZ * KDIM * sizeof(__bf16)); // 2048x640
    __bf16* Act     = (__bf16*)take((size_t)B_SZ * KDIM * sizeof(__bf16));     // 128x640
    float*  Hs      = (float*) take((size_t)M_STEPS * B_SZ * H_SZ * sizeof(float));
    float*  Cs      = (float*) take((size_t)M_STEPS * B_SZ * H_SZ * sizeof(float));
    float*  h_state = (float*) take((size_t)B_SZ * H_SZ * sizeof(float));
    float*  c_state = (float*) take((size_t)B_SZ * H_SZ * sizeof(float));
    float*  hdot    = (float*) take((size_t)M_STEPS * B_SZ * sizeof(float));
    float*  ponder  = (float*) take(sizeof(float));

    build_wcat_kernel<<<4 * H_SZ, 128, 0, stream>>>(W_ih, W_hh, WcatT);
    init_state_kernel<<<(B_SZ * H_SZ + 255) / 256, 256, 0, stream>>>(h_state, c_state, hdot, ponder);

    const size_t slab = (size_t)B_SZ * H_SZ;
    for (int t = 0; t < T_SZ; ++t) {
        for (int n = 0; n < M_STEPS; ++n) {
            const float* h_src = (n == 0) ? h_state : Hs + (size_t)(n - 1) * slab;
            const float* c_src = (n == 0) ? c_state : Cs + (size_t)(n - 1) * slab;
            pack_act_kernel<<<B_SZ, 128, 0, stream>>>(x, h_src, Act, hdot + n * B_SZ,
                                                      t, (n == 0) ? 1.0f : 0.0f, (n == 0) ? 1 : 0);
            lstm_step_kernel<<<dim3(B_SZ / 16, H_SZ / 16), 256, 0, stream>>>(
                Act, WcatT, bias, c_src, w_halt,
                Hs + (size_t)n * slab, Cs + (size_t)n * slab, hdot + n * B_SZ);
        }
        act_combine_kernel<<<(B_SZ * H_SZ) / 256, 256, 0, stream>>>(
            Hs, Cs, hdot, b_halt, h_state, c_state, ponder);
    }
    fc_out_kernel<<<B_SZ, 256, 0, stream>>>(h_state, W_fc, b_fc, ponder, out);
}
